// PointConv_81587198754960
// MI455X (gfx1250) — matrix operand order repaired and checked by reading, not compiled, not run on hardware
//
#include <hip/hip_runtime.h>

#define N_NODES 50000
#define N_EDGES 1600000
#define C 32
#define A_DIM 4
#define R_DIM 8
#define H_DIM 64
#define WPB 4   // waves per block in edge kernel

typedef __attribute__((ext_vector_type(16))) _Float16 v16h;
typedef __attribute__((ext_vector_type(8)))  _Float16 v8h;
typedef __attribute__((ext_vector_type(8)))  float    v8f;

// ---------------- kernel 1: node prep (lin1 on s and v) ----------------
__global__ void node_prep_kernel(const float* __restrict__ nf,
                                 const float* __restrict__ w0,
                                 const float* __restrict__ w1,
                                 float* __restrict__ s_arr,
                                 float* __restrict__ v_arr) {
  __shared__ float l0[C][C];
  __shared__ float l1[C][C];
  int tid = threadIdx.x;
  for (int i = tid; i < C * C; i += 256) {
    l0[i / C][i % C] = w0[i];
    l1[i / C][i % C] = w1[i];
  }
  __syncthreads();
  int n = blockIdx.x * 256 + tid;
  if (n >= N_NODES) return;
  const float inv_sC = 0.17677669529663687f; // 1/sqrt(32)
  const float* row = nf + (size_t)n * (4 * C);
  float sin_[C];
  #pragma unroll
  for (int u = 0; u < C; ++u) sin_[u] = row[u];
  float* srow = s_arr + (size_t)n * C;
  for (int w = 0; w < C; ++w) {
    float acc = 0.f;
    #pragma unroll
    for (int u = 0; u < C; ++u) acc += sin_[u] * l0[u][w];
    srow[w] = acc * inv_sC;
  }
  float* vrow = v_arr + (size_t)n * (3 * C);
  for (int i = 0; i < 3; ++i) {
    float vin[C];
    #pragma unroll
    for (int u = 0; u < C; ++u) vin[u] = row[C + u * 3 + i];
    for (int w = 0; w < C; ++w) {
      float acc = 0.f;
      #pragma unroll
      for (int u = 0; u < C; ++u) acc += vin[u] * l1[u][w];
      vrow[w * 3 + i] = acc * inv_sC;
    }
  }
}

// ---------------- kernel 2: fused edge stage (WMMA MLP + messages) ----------------
// B-fragment element mapping (this code's k order, 16-bit B 32x16):
//   lane L: half = L/16, col = tile*16 + L%16; element t (0..15) <-> K = half*16 + t
// A-fragment element mapping (16-bit A 16x32):
//   lane L: row = L%16; elements 0..7  <-> K = half*16 + 0..7
//                       elements 8..15 <-> K = half*16 + 8..15   (contiguous!)
__global__ void edge_kernel(const float* __restrict__ emb,
                            const float* __restrict__ eattr,
                            const int*   __restrict__ eidx,
                            const float* __restrict__ mlp_w0,
                            const float* __restrict__ mlp_w1,
                            const float* __restrict__ s_arr,
                            const float* __restrict__ v_arr,
                            float* __restrict__ msg_s,
                            float* __restrict__ msg_v) {
  __shared__ __align__(32) _Float16 w0frag[4][32][16];     // 4 KB  pre-baked GEMM1 B frags
  __shared__ __align__(32) _Float16 w1frag[2][8][32][16];  // 16 KB pre-baked GEMM2 B frags
  __shared__ __align__(16) _Float16 hst[WPB][16][H_DIM];   // 8 KB  per-wave h staging
  __shared__ float wst[WPB][16][4 * C];                    // 32 KB per-wave w staging
  const int tid = threadIdx.x;

  // bake GEMM1 B fragments (K>=8 rows are zero padding)
  for (int i = tid; i < 4 * 32 * 16; i += WPB * 32) {
    int t = i & 15, ln = (i >> 4) & 31, j = i >> 9;
    int K = (ln >> 4) * 16 + t;
    int n = j * 16 + (ln & 15);
    w0frag[j][ln][t] = (K < R_DIM) ? (_Float16)mlp_w0[K * H_DIM + n] : (_Float16)0.f;
  }
  // bake GEMM2 B fragments
  for (int i = tid; i < 2 * 8 * 32 * 16; i += WPB * 32) {
    int t = i & 15, ln = (i >> 4) & 31, j = (i >> 9) & 7, c2 = i >> 12;
    int K = c2 * 32 + (ln >> 4) * 16 + t;
    int n = j * 16 + (ln & 15);
    w1frag[c2][j][ln][t] = (_Float16)mlp_w1[K * (4 * C) + n];
  }
  __syncthreads();

  const int wave = tid >> 5;
  const int lane = tid & 31;
  const int half = lane >> 4;
  const int m16  = lane & 15;
  const long e_base = ((long)blockIdx.x * WPB + wave) * 16;

  // prefetch epilogue streams while the MLP runs
  if (e_base < N_EDGES) {
    __builtin_prefetch(eattr + e_base * 4, 0, 3);
    __builtin_prefetch(eidx + e_base, 0, 3);
    __builtin_prefetch(eidx + (long)N_EDGES + e_base, 0, 3);
  }

  // ---- GEMM1: h_pre = emb(16x8, K zero-padded to 32) @ W0(8x64) ----
  v16h aF = {};
  if (half == 0) {            // K<8 lives in half 0, elements 0..7 (K == element idx)
    long e = e_base + m16;
    if (e < N_EDGES) {
      const float4* er = (const float4*)(emb + e * R_DIM);
      float4 e0 = er[0], e1 = er[1];
      aF[0] = (_Float16)e0.x; aF[1] = (_Float16)e0.y;
      aF[2] = (_Float16)e0.z; aF[3] = (_Float16)e0.w;
      aF[4] = (_Float16)e1.x; aF[5] = (_Float16)e1.y;
      aF[6] = (_Float16)e1.z; aF[7] = (_Float16)e1.w;
    }
  }
  const v8f cz = {};
  const float inv_sR = 0.35355339059327373f; // 1/sqrt(8)
  #pragma unroll
  for (int j = 0; j < 4; ++j) {
    v16h bF = *(const v16h*)&w0frag[j][lane][0];           // 2x ds_load_b128
    v8f d = __builtin_amdgcn_wmma_f32_16x16x32_f16(false, aF, false, bF,
                                                   (short)0, cz, false, false);
    #pragma unroll
    for (int r = 0; r < 8; ++r) {      // D: row = r + 8*half, col = j*16 + m16
      float x = d[r] * inv_sR;
      float h = x * __builtin_amdgcn_rcpf(1.f + __expf(-x));   // fast SiLU
      hst[wave][r + 8 * half][j * 16 + m16] = (_Float16)h;
    }
  }

  // ---- GEMM2: w = h(16x64) @ W1(64x128), K = 2 chunks of 32 ----
  v8f acc[8];
  #pragma unroll
  for (int j = 0; j < 8; ++j) acc[j] = cz;
  #pragma unroll
  for (int c2 = 0; c2 < 2; ++c2) {
    v8h alo = *(const v8h*)&hst[wave][m16][c2 * 32 + half * 16];      // ds_load_b128
    v8h ahi = *(const v8h*)&hst[wave][m16][c2 * 32 + half * 16 + 8];  // ds_load_b128
    v16h a2 = __builtin_shufflevector(alo, ahi, 0, 1, 2, 3, 4, 5, 6, 7,
                                      8, 9, 10, 11, 12, 13, 14, 15);
    #pragma unroll
    for (int j = 0; j < 8; ++j) {
      v16h b2 = *(const v16h*)&w1frag[c2][j][lane][0];     // 2x ds_load_b128
      acc[j] = __builtin_amdgcn_wmma_f32_16x16x32_f16(false, a2, false, b2,
                                                      (short)0, acc[j], false, false);
    }
  }
  const float inv_sH = 0.125f; // 1/sqrt(64)
  #pragma unroll
  for (int j = 0; j < 8; ++j)
    #pragma unroll
    for (int r = 0; r < 8; ++r)
      wst[wave][r + 8 * half][j * 16 + m16] = acc[j][r] * inv_sH;

  // ---- per-edge messages + atomic scatter; lane == channel ----
  const float inv_s2 = 0.7071067811865476f;
  const float inv_s3 = 0.5773502691896258f;
  const int c = lane;
  for (int m = 0; m < 16; ++m) {
    long e = e_base + m;
    if (e >= N_EDGES) break;
    int src = eidx[e];
    int dst = eidx[(long)N_EDGES + e];
    float4 ea = *(const float4*)(eattr + e * 4);
    float a0 = ea.x, a1x = ea.y, a1y = ea.z, a1z = ea.w;
    float w1c = wst[wave][m][c];
    float w2c = wst[wave][m][C + c];
    float w3c = wst[wave][m][2 * C + c];
    float w4c = wst[wave][m][3 * C + c];
    const float* sp = s_arr + (size_t)src * C;
    const float* vp = v_arr + (size_t)src * (3 * C) + c * 3;
    float se  = sp[c];
    float vex = vp[0], vey = vp[1], vez = vp[2];
    float dot = vex * a1x + vey * a1y + vez * a1z;
    float ms  = (w1c * se * a0 + w4c * dot * inv_s3) * inv_s2;
    float mvx = (w2c * se * a1x + w3c * vex * a0) * inv_s2;
    float mvy = (w2c * se * a1y + w3c * vey * a0) * inv_s2;
    float mvz = (w2c * se * a1z + w3c * vez * a0) * inv_s2;
    atomicAdd(&msg_s[(size_t)dst * C + c], ms);
    float* mvp = msg_v + (size_t)dst * (3 * C) + (size_t)c * 3;
    atomicAdd(mvp + 0, mvx);
    atomicAdd(mvp + 1, mvy);
    atomicAdd(mvp + 2, mvz);
  }
}

// ---------------- kernel 3: node output (lin2 + skip connection) ----------------
__global__ void node_out_kernel(const float* __restrict__ nf,
                                const float* __restrict__ nattr,
                                const float* __restrict__ lin2_w0,
                                const float* __restrict__ lin2_w1,
                                const float* __restrict__ sc_w0,
                                const float* __restrict__ sc_w1,
                                const float* __restrict__ msg_s,
                                const float* __restrict__ msg_v,
                                float* __restrict__ out) {
  __shared__ float l0[C][C], l1[C][C];
  __shared__ float sw0[C][A_DIM][C], sw1[C][A_DIM][C];
  int tid = threadIdx.x;
  for (int i = tid; i < C * C; i += 256) {
    l0[i / C][i % C] = lin2_w0[i];
    l1[i / C][i % C] = lin2_w1[i];
  }
  for (int i = tid; i < C * A_DIM * C; i += 256) {
    int u = i / (A_DIM * C); int rem = i % (A_DIM * C);
    sw0[u][rem / C][rem % C] = sc_w0[i];
    sw1[u][rem / C][rem % C] = sc_w1[i];
  }
  __syncthreads();
  int n = blockIdx.x * 256 + tid;
  if (n >= N_NODES) return;
  const float inv_sC  = 0.17677669529663687f; // 1/sqrt(32)
  const float inv_deg = 0.17677669529663687f; // 1/sqrt(AVG_NEIGH=32)
  const float inv_fc  = 0.08838834764831845f; // 1/sqrt(C*A=128)
  float at[A_DIM];
  #pragma unroll
  for (int a = 0; a < A_DIM; ++a) at[a] = nattr[(size_t)n * A_DIM + a];
  const float* row = nf + (size_t)n * (4 * C);
  float* orow = out + (size_t)n * (4 * C);
  {
    float ms[C], si[C];
    #pragma unroll
    for (int u = 0; u < C; ++u) {
      ms[u] = msg_s[(size_t)n * C + u] * inv_deg;
      si[u] = row[u];
    }
    for (int w = 0; w < C; ++w) {
      float acc = 0.f, accsc = 0.f;
      #pragma unroll 4
      for (int u = 0; u < C; ++u) {
        acc += ms[u] * l0[u][w];
        float t = at[0] * sw0[u][0][w] + at[1] * sw0[u][1][w]
                + at[2] * sw0[u][2][w] + at[3] * sw0[u][3][w];
        accsc += si[u] * t;
      }
      orow[w] = acc * inv_sC + accsc * inv_fc;
    }
  }
  for (int i = 0; i < 3; ++i) {
    float mv[C], vi[C];
    #pragma unroll
    for (int u = 0; u < C; ++u) {
      mv[u] = msg_v[(size_t)n * 3 * C + u * 3 + i] * inv_deg;
      vi[u] = row[C + u * 3 + i];
    }
    for (int w = 0; w < C; ++w) {
      float acc = 0.f, accsc = 0.f;
      #pragma unroll 4
      for (int u = 0; u < C; ++u) {
        acc += mv[u] * l1[u][w];
        float t = at[0] * sw1[u][0][w] + at[1] * sw1[u][1][w]
                + at[2] * sw1[u][2][w] + at[3] * sw1[u][3][w];
        accsc += vi[u] * t;
      }
      orow[C + w * 3 + i] = acc * inv_sC + accsc * inv_fc;
    }
  }
}

extern "C" void kernel_launch(void* const* d_in, const int* in_sizes, int n_in,
                              void* d_out, int out_size, void* d_ws, size_t ws_size,
                              hipStream_t stream) {
  const float* node_feats = (const float*)d_in[0];
  const float* node_attrs = (const float*)d_in[1];
  const float* edge_attrs = (const float*)d_in[2];
  const float* edge_emb   = (const float*)d_in[3];
  const int*   edge_index = (const int*)d_in[4];
  const float* lin1_w0 = (const float*)d_in[5];
  const float* lin1_w1 = (const float*)d_in[6];
  const float* mlp_w0  = (const float*)d_in[7];
  const float* mlp_w1  = (const float*)d_in[8];
  const float* lin2_w0 = (const float*)d_in[9];
  const float* lin2_w1 = (const float*)d_in[10];
  const float* sc_w0   = (const float*)d_in[11];
  const float* sc_w1   = (const float*)d_in[12];

  // workspace layout (floats): s(N*C) | v(N*3C) | msg_s(N*C) | msg_v(N*3C)
  float* s_arr = (float*)d_ws;
  float* v_arr = s_arr + (size_t)N_NODES * C;
  float* msg_s = v_arr + (size_t)N_NODES * 3 * C;
  float* msg_v = msg_s + (size_t)N_NODES * C;

  hipMemsetAsync(msg_s, 0, (size_t)N_NODES * 4 * C * sizeof(float), stream);

  node_prep_kernel<<<(N_NODES + 255) / 256, 256, 0, stream>>>(
      node_feats, lin1_w0, lin1_w1, s_arr, v_arr);

  int eblocks = (N_EDGES + WPB * 16 - 1) / (WPB * 16);
  edge_kernel<<<eblocks, WPB * 32, 0, stream>>>(
      edge_emb, edge_attrs, edge_index, mlp_w0, mlp_w1,
      s_arr, v_arr, msg_s, msg_v);

  node_out_kernel<<<(N_NODES + 255) / 256, 256, 0, stream>>>(
      node_feats, node_attrs, lin2_w0, lin2_w1, sc_w0, sc_w1,
      msg_s, msg_v, (float*)d_out);
}
